// LancetBlockConfigurable_7894149890760
// MI455X (gfx1250) — matrix operand order repaired and checked by reading, not compiled, not run on hardware
//
#include <hip/hip_runtime.h>
#include <hip/hip_bf16.h>

// ---------------------------------------------------------------------------
// CDNA5 / gfx1250: LayerNorm + MHA(+residual) + MoE FFN.
// All GEMMs via v_wmma_f32_16x16x32_bf16 (fp32 -> bf16 at LDS staging, fp32
// accum). Flash attention (online softmax). Gating/top-k discarded in the
// reference -> skipped.
// Round 2 changes: packed bf16 LDS staging (dword/8B stores instead of
// per-element b16 stores) and software-pipelined, double-buffered GEMM main
// loop + register-prefetched attention K/V staging so global loads overlap
// the WMMA work.
// ---------------------------------------------------------------------------

typedef __attribute__((ext_vector_type(16))) __bf16 v16bf;
typedef __attribute__((ext_vector_type(8)))  __bf16 v8bf;
typedef __attribute__((ext_vector_type(8)))  float  v8f;

__device__ __forceinline__ __bf16 f2bf(float x) {
  unsigned u = __builtin_bit_cast(unsigned, x);
  unsigned r = u + 0x7FFFu + ((u >> 16) & 1u);
  unsigned short s = (unsigned short)(r >> 16);
  return __builtin_bit_cast(__bf16, s);
}

// pack two fp32 -> two bf16 in one dword (lo = a, hi = b), RNE
__device__ __forceinline__ unsigned bfpack2(float a, float b) {
#if __has_builtin(__builtin_amdgcn_cvt_pk_bf16_f32)
  typedef __attribute__((ext_vector_type(2))) __bf16 v2bf;
  v2bf p = __builtin_amdgcn_cvt_pk_bf16_f32(a, b);
  return __builtin_bit_cast(unsigned, p);
#else
  unsigned ua = __builtin_bit_cast(unsigned, a);
  unsigned ub = __builtin_bit_cast(unsigned, b);
  unsigned sa = ua + 0x7FFFu + ((ua >> 16) & 1u);
  unsigned sb = ub + 0x7FFFu + ((ub >> 16) & 1u);
  return (sa >> 16) | (sb & 0xFFFF0000u);
#endif
}

__device__ __forceinline__ v8f v8f_zero() {
  v8f z = {0.f, 0.f, 0.f, 0.f, 0.f, 0.f, 0.f, 0.f};
  return z;
}

__device__ __forceinline__ v16bf frag16(const __bf16* p0, const __bf16* p1) {
  v8bf lo = *(const v8bf*)p0;
  v8bf hi = *(const v8bf*)p1;
  v16bf f;
#pragma unroll
  for (int e = 0; e < 8; ++e) { f[e] = lo[e]; f[e + 8] = hi[e]; }
  return f;
}

// store float4 as 4 packed bf16 (8 bytes) to LDS
__device__ __forceinline__ void st_bf4(__bf16* dst, float4 v) {
  uint2 u;
  u.x = bfpack2(v.x, v.y);
  u.y = bfpack2(v.z, v.w);
  *(uint2*)dst = u;
}

// ---------------------------------------------------------------------------
// LayerNorm (biased variance). One block per row, 256 threads.
// ---------------------------------------------------------------------------
__global__ __launch_bounds__(256)
void ln_kernel(const float* __restrict__ x, const float* __restrict__ g,
               const float* __restrict__ bta, float* __restrict__ xn, int D) {
  long long row = blockIdx.x;
  const float* xr = x + row * D;
  float vals[4];
  float s = 0.f, s2 = 0.f;
#pragma unroll
  for (int i = 0; i < 4; ++i) {
    float v = xr[threadIdx.x + i * 256];
    vals[i] = v; s += v; s2 += v * v;
  }
#pragma unroll
  for (int off = 1; off < 32; off <<= 1) {
    s  += __shfl_xor(s,  off, 32);
    s2 += __shfl_xor(s2, off, 32);
  }
  __shared__ float red[2][8];
  int w = threadIdx.x >> 5, lane = threadIdx.x & 31;
  if (lane == 0) { red[0][w] = s; red[1][w] = s2; }
  __syncthreads();
  float ts = 0.f, ts2 = 0.f;
#pragma unroll
  for (int i = 0; i < 8; ++i) { ts += red[0][i]; ts2 += red[1][i]; }
  float mu  = ts / (float)D;
  float var = ts2 / (float)D - mu * mu;
  float inv = rsqrtf(var + 1e-5f);
#pragma unroll
  for (int i = 0; i < 4; ++i) {
    int c = threadIdx.x + i * 256;
    xn[row * D + c] = (vals[i] - mu) * inv * g[c] + bta[c];
  }
}

// ---------------------------------------------------------------------------
// WMMA GEMM: C = op(A @ B (+bias) (gelu)) (+resid). fp32 in/out, bf16 math.
// Block tile 128x128x32; 8 waves, wave tile 32x64 (2x4 WMMA tiles).
// Software-pipelined: double-buffered LDS; tile k+1 global loads issue
// before tile k's WMMAs; converted/stored to the alternate buffer after.
// blockIdx.z slices (microbatch,expert) via *zs strides; B/bias use z%bmod.
// Requires M,N % 128 == 0, K % 32 == 0.
// ---------------------------------------------------------------------------
template <bool BIAS, bool GELU, bool RESID>
__global__ __launch_bounds__(256)
void gemm_bf16w(const float* __restrict__ A, const float* __restrict__ Bm,
                const float* __restrict__ bias, const float* __restrict__ resid,
                float* __restrict__ C, int M, int N, int K,
                long long azs, long long bzs, long long biaszs, long long czs,
                int bmod) {
  constexpr int SA = 40;  // A row stride (halfs): 32 + 8 pad, 16B-aligned chunks
  constexpr int SB = 40;  // Bs transposed: Bs[n][k]
  __shared__ __bf16 As[2][128 * SA];
  __shared__ __bf16 Bs[2][128 * SB];

  int z  = blockIdx.z;
  int zb = (bmod > 0) ? (z % bmod) : z;
  A  += (long long)z  * azs;
  Bm += (long long)zb * bzs;
  if constexpr (BIAS) bias += (long long)zb * biaszs;
  C  += (long long)z * czs;
  if constexpr (RESID) resid += (long long)z * czs;

  const int m0 = blockIdx.y * 128;
  const int n0 = blockIdx.x * 128;
  const int t = threadIdx.x;
  const int lane = t & 31;
  const int w  = t >> 5;
  const int wm = w & 3;
  const int wn = w >> 2;
  const int lh  = (lane >> 4) & 1;
  const int l15 = lane & 15;

  // staging geometry (per thread)
  const int arow = t >> 3;          // + 32*i
  const int acol = (t & 7) << 2;
  const int bk   = t >> 5;          // + 8*i
  const int bn   = (t & 31) << 2;

  float4 ra[4], rb[4];
  auto loadAB = [&](int k0) {
#pragma unroll
    for (int i = 0; i < 4; ++i)
      ra[i] = *(const float4*)(A + (long long)(m0 + arow + 32 * i) * K + k0 + acol);
#pragma unroll
    for (int i = 0; i < 4; ++i)
      rb[i] = *(const float4*)(Bm + (long long)(k0 + bk + 8 * i) * N + n0 + bn);
  };
  auto storeAB = [&](int buf) {
#pragma unroll
    for (int i = 0; i < 4; ++i)
      st_bf4(&As[buf][(arow + 32 * i) * SA + acol], ra[i]);
#pragma unroll
    for (int i = 0; i < 4; ++i) {  // transpose scatter: Bs[n][k]
      int kk = bk + 8 * i;
      Bs[buf][(bn + 0) * SB + kk] = f2bf(rb[i].x);
      Bs[buf][(bn + 1) * SB + kk] = f2bf(rb[i].y);
      Bs[buf][(bn + 2) * SB + kk] = f2bf(rb[i].z);
      Bs[buf][(bn + 3) * SB + kk] = f2bf(rb[i].w);
    }
  };

  v8f acc[2][4];
#pragma unroll
  for (int mt = 0; mt < 2; ++mt)
#pragma unroll
    for (int nt = 0; nt < 4; ++nt) acc[mt][nt] = v8f_zero();

  const int KT = K >> 5;
  loadAB(0);
  storeAB(0);
  __syncthreads();

  for (int kt = 0; kt < KT; ++kt) {
    const int cur = kt & 1;
    if (kt + 1 < KT) loadAB((kt + 1) << 5);  // vmem in flight during WMMAs

    v16bf af[2], bf[4];
#pragma unroll
    for (int mt = 0; mt < 2; ++mt) {
      int m = wm * 32 + mt * 16 + l15;
      af[mt] = frag16(&As[cur][m * SA + lh * 8], &As[cur][m * SA + 16 + lh * 8]);
    }
#pragma unroll
    for (int nt = 0; nt < 4; ++nt) {
      int n = wn * 64 + nt * 16 + l15;
      bf[nt] = frag16(&Bs[cur][n * SB + lh * 8], &Bs[cur][n * SB + 16 + lh * 8]);
    }
#pragma unroll
    for (int mt = 0; mt < 2; ++mt)
#pragma unroll
      for (int nt = 0; nt < 4; ++nt)
        acc[mt][nt] = __builtin_amdgcn_wmma_f32_16x16x32_bf16(
            false, af[mt], false, bf[nt], (short)0, acc[mt][nt], false, false);

    if (kt + 1 < KT) storeAB(cur ^ 1);
    __syncthreads();
  }

  // epilogue: C frag layout: col = base + (lane&15), row = r + 8*(lane>=16)
#pragma unroll
  for (int mt = 0; mt < 2; ++mt) {
#pragma unroll
    for (int nt = 0; nt < 4; ++nt) {
      int n = n0 + wn * 64 + nt * 16 + l15;
      float bv = 0.f;
      if constexpr (BIAS) bv = bias[n];
#pragma unroll
      for (int r = 0; r < 8; ++r) {
        int m = m0 + wm * 32 + mt * 16 + r + lh * 8;
        float v = acc[mt][nt][r];
        if constexpr (BIAS) v += bv;
        if constexpr (GELU) v = 0.5f * v * (1.f + erff(v * 0.70710678118654752f));
        if constexpr (RESID) v += resid[(long long)m * N + n];
        C[(long long)m * N + n] = v;
      }
    }
  }
}

// ---------------------------------------------------------------------------
// Flash attention: block = (b*H+h, 128-query block); hd=32 -> one WMMA
// K-step for Q@K^T. K/V tiles register-prefetched one block ahead.
// ---------------------------------------------------------------------------
__global__ __launch_bounds__(256)
void attn_kernel(const float* __restrict__ Q, const float* __restrict__ Kb,
                 const float* __restrict__ Vb, float* __restrict__ O,
                 int B, int S, int D, int H) {
  constexpr int SQ = 40, SK = 40, SV = 136, SP = 136;
  __shared__ __bf16 Qs[128 * SQ];
  __shared__ __bf16 Ks[128 * SK];
  __shared__ __bf16 Vs[32 * SV];      // V transposed: Vs[d][kk]
  __shared__ __bf16 Ps[8 * 16 * SP];  // per-wave P slab (16 x 128)

  const int hd = 32;
  const float scale = 0.17677669529663687f;  // 1/sqrt(32)
  int bh = blockIdx.y;
  int b = bh / H, h = bh % H;
  int q0 = blockIdx.x * 128;
  const float* Qp = Q  + (long long)b * S * D + h * hd;
  const float* Kp = Kb + (long long)b * S * D + h * hd;
  const float* Vp = Vb + (long long)b * S * D + h * hd;

  int t = threadIdx.x, lane = t & 31, w = t >> 5;
  int lh = (lane >> 4) & 1, l15 = lane & 15;
  const int srow = t >> 3;         // + 32*i
  const int scol = (t & 7) << 2;

  // stage Q block (pre-scaled), packed stores
#pragma unroll
  for (int i = 0; i < 4; ++i) {
    float4 v = *(const float4*)(Qp + (long long)(q0 + srow + 32 * i) * D + scol);
    v.x *= scale; v.y *= scale; v.z *= scale; v.w *= scale;
    st_bf4(&Qs[(srow + 32 * i) * SQ + scol], v);
  }
  __syncthreads();
  v16bf qf;
  {
    int m = w * 16 + l15;
    qf = frag16(&Qs[m * SQ + lh * 8], &Qs[m * SQ + 16 + lh * 8]);
  }

  float4 rk[4], rv[4];
  auto loadKV = [&](int kb) {
#pragma unroll
    for (int i = 0; i < 4; ++i)
      rk[i] = *(const float4*)(Kp + (long long)(kb + srow + 32 * i) * D + scol);
#pragma unroll
    for (int i = 0; i < 4; ++i)
      rv[i] = *(const float4*)(Vp + (long long)(kb + srow + 32 * i) * D + scol);
  };
  auto storeKV = [&]() {
#pragma unroll
    for (int i = 0; i < 4; ++i)
      st_bf4(&Ks[(srow + 32 * i) * SK + scol], rk[i]);
#pragma unroll
    for (int i = 0; i < 4; ++i) {  // V transpose scatter
      int r = srow + 32 * i;
      Vs[(scol + 0) * SV + r] = f2bf(rv[i].x);
      Vs[(scol + 1) * SV + r] = f2bf(rv[i].y);
      Vs[(scol + 2) * SV + r] = f2bf(rv[i].z);
      Vs[(scol + 3) * SV + r] = f2bf(rv[i].w);
    }
  };

  float mrow[8], lrow[8];
#pragma unroll
  for (int r = 0; r < 8; ++r) { mrow[r] = -1e30f; lrow[r] = 0.f; }
  v8f oacc[2];
  oacc[0] = v8f_zero(); oacc[1] = v8f_zero();

  loadKV(0);
  for (int kb = 0; kb < S; kb += 128) {
    __syncthreads();   // prev iteration done reading Ks/Vs
    storeKV();
    if (kb + 128 < S) loadKV(kb + 128);  // vmem in flight during compute
    __syncthreads();

    // scores: 16 (rows) x 128 (keys) per wave, single K-step
    v8f sc[8];
#pragma unroll
    for (int nt = 0; nt < 8; ++nt) {
      int n = nt * 16 + l15;
      v16bf kf = frag16(&Ks[n * SK + lh * 8], &Ks[n * SK + 16 + lh * 8]);
      sc[nt] = __builtin_amdgcn_wmma_f32_16x16x32_bf16(
          false, qf, false, kf, (short)0, v8f_zero(), false, false);
    }

    // online softmax; row = r + 8*lh spans 16 lanes of a half-wave
    float mnew[8], al[8];
#pragma unroll
    for (int r = 0; r < 8; ++r) {
      float v = sc[0][r];
#pragma unroll
      for (int nt = 1; nt < 8; ++nt) v = fmaxf(v, sc[nt][r]);
#pragma unroll
      for (int off = 1; off < 16; off <<= 1) v = fmaxf(v, __shfl_xor(v, off, 32));
      mnew[r] = fmaxf(mrow[r], v);
      al[r] = __expf(mrow[r] - mnew[r]);
      mrow[r] = mnew[r];
    }
    float rs[8];
#pragma unroll
    for (int r = 0; r < 8; ++r) rs[r] = 0.f;
#pragma unroll
    for (int nt = 0; nt < 8; ++nt)
#pragma unroll
      for (int r = 0; r < 8; ++r) {
        float p = __expf(sc[nt][r] - mnew[r]);
        sc[nt][r] = p;
        rs[r] += p;
      }
#pragma unroll
    for (int r = 0; r < 8; ++r) {
      float v = rs[r];
#pragma unroll
      for (int off = 1; off < 16; off <<= 1) v += __shfl_xor(v, off, 32);
      lrow[r] = lrow[r] * al[r] + v;
      oacc[0][r] *= al[r];
      oacc[1][r] *= al[r];
    }

    // P -> per-wave LDS slab (acc layout -> A-fragment layout)
#pragma unroll
    for (int nt = 0; nt < 8; ++nt)
#pragma unroll
      for (int r = 0; r < 8; ++r)
        Ps[(w * 16 + r + lh * 8) * SP + nt * 16 + l15] = f2bf(sc[nt][r]);
    asm volatile("s_wait_dscnt 0" ::: "memory");  // same-wave LDS RAW

    // O += P @ V  (128 keys -> 4 WMMA K-steps)
#pragma unroll
    for (int kc = 0; kc < 4; ++kc) {
      int m = w * 16 + l15;
      v16bf pf = frag16(&Ps[m * SP + kc * 32 + lh * 8],
                        &Ps[m * SP + kc * 32 + 16 + lh * 8]);
#pragma unroll
      for (int nt = 0; nt < 2; ++nt) {
        int n = nt * 16 + l15;
        v16bf vf = frag16(&Vs[n * SV + kc * 32 + lh * 8],
                          &Vs[n * SV + kc * 32 + 16 + lh * 8]);
        oacc[nt] = __builtin_amdgcn_wmma_f32_16x16x32_bf16(
            false, pf, false, vf, (short)0, oacc[nt], false, false);
      }
    }
  }

  // normalize and write o[b, q, h, d] (fp32)
#pragma unroll
  for (int nt = 0; nt < 2; ++nt) {
    int d = nt * 16 + l15;
#pragma unroll
    for (int r = 0; r < 8; ++r) {
      int qrow = q0 + w * 16 + r + lh * 8;
      O[(long long)(b * S + qrow) * D + h * hd + d] = oacc[nt][r] / lrow[r];
    }
  }
}

// ---------------------------------------------------------------------------
// Launcher
// ---------------------------------------------------------------------------
extern "C" void kernel_launch(void* const* d_in, const int* in_sizes, int n_in,
                              void* d_out, int out_size, void* d_ws, size_t ws_size,
                              hipStream_t stream) {
  (void)in_sizes; (void)n_in; (void)out_size; (void)ws_size;
  const float* x    = (const float*)d_in[0];
  const float* ln_g = (const float*)d_in[1];
  const float* ln_b = (const float*)d_in[2];
  const float* wq   = (const float*)d_in[3];
  const float* wk   = (const float*)d_in[4];
  const float* wv   = (const float*)d_in[5];
  const float* wo   = (const float*)d_in[6];
  // d_in[7] = gate_w: gating result discarded in reference -> skipped
  const float* w1   = (const float*)d_in[8];
  const float* b1   = (const float*)d_in[9];
  const float* w2   = (const float*)d_in[10];
  const float* b2   = (const float*)d_in[11];

  const int B = 8, S = 1024, D = 1024, E = 8, Fd = 4096, H = 32, MBc = 2;
  const long long T = (long long)B * S;      // 8192 tokens
  const int tpe = (int)(T / (MBc * E));      // 512 tokens/expert/microbatch

  // workspace (fp32): xn | q | k | v | h ; o reuses xn, attn+res reuses q
  float* xn = (float*)d_ws;
  float* qb = xn + T * D;
  float* kb = qb + T * D;
  float* vb = kb + T * D;
  float* hb = vb + T * D;   // T * Fd
  float* ob = xn;
  float* ar = qb;

  ln_kernel<<<(int)T, 256, 0, stream>>>(x, ln_g, ln_b, xn, D);

  dim3 gqkv(D / 128, (int)(T / 128), 1);
  gemm_bf16w<false, false, false><<<gqkv, 256, 0, stream>>>(
      xn, wq, nullptr, nullptr, qb, (int)T, D, D, 0, 0, 0, 0, 0);
  gemm_bf16w<false, false, false><<<gqkv, 256, 0, stream>>>(
      xn, wk, nullptr, nullptr, kb, (int)T, D, D, 0, 0, 0, 0, 0);
  gemm_bf16w<false, false, false><<<gqkv, 256, 0, stream>>>(
      xn, wv, nullptr, nullptr, vb, (int)T, D, D, 0, 0, 0, 0, 0);

  dim3 ga(S / 128, B * H, 1);
  attn_kernel<<<ga, 256, 0, stream>>>(qb, kb, vb, ob, B, S, D, H);

  gemm_bf16w<false, false, true><<<gqkv, 256, 0, stream>>>(
      ob, wo, nullptr, x, ar, (int)T, D, D, 0, 0, 0, 0, 0);

  dim3 g1(Fd / 128, tpe / 128, MBc * E);
  gemm_bf16w<true, true, false><<<g1, 256, 0, stream>>>(
      ar, w1, b1, nullptr, hb, tpe, Fd, D,
      (long long)tpe * D, (long long)D * Fd, (long long)Fd,
      (long long)tpe * Fd, E);

  dim3 g2(D / 128, tpe / 128, MBc * E);
  gemm_bf16w<true, false, false><<<g2, 256, 0, stream>>>(
      hb, w2, b2, nullptr, (float*)d_out, tpe, D, Fd,
      (long long)tpe * Fd, (long long)Fd * D, (long long)D,
      (long long)tpe * D, E);
}
